// ReciprocalSpectralKernel3D_66881230733956
// MI455X (gfx1250) — matrix operand order repaired and checked by reading, not compiled, not run on hardware
//
#include <hip/hip_runtime.h>
#include <math.h>

// ---------------------------------------------------------------------------
// ReciprocalSpectralKernel3D for MI455X (gfx1250, wave32, WMMA)
//
// Pipeline (all on `stream`, scratch in d_ws):
//   k_setup  : per-graph 3x3 inverse + |det|, graph start offsets (binary search)
//   k_filter : k-space features -> MLP(3->64->64->1) -> radial weight w[b,k]
//              64x64 hidden layer runs on v_wmma_f32_16x16x32_f16 (f16 A/B, f32 acc)
//   k_struct : structure factors S_cos/S_sin[b,k,c] (one block per graph,
//              register accumulators, no atomics -- batch is sorted)
//   k_energy : per-atom reciprocal-space energy (L2-resident S arrays)
// ---------------------------------------------------------------------------

typedef __attribute__((ext_vector_type(16))) _Float16 v16h;
typedef __attribute__((ext_vector_type(8)))  _Float16 v8h;
typedef __attribute__((ext_vector_type(8)))  float    v8f;

static constexpr int   N_ATOMS = 8192;
static constexpr int   NGRAPH  = 32;
static constexpr int   NCH     = 4;
static constexpr int   HID     = 64;
static constexpr int   NK      = 728;     // 9^3 - 1 (k=0 excluded)
static constexpr float KFLOOR  = 1e-6f;
#define PI_F    3.14159265358979323846f
#define TWO_PI  6.28318530717958647692f

// workspace layout (float offsets from d_ws base); total ~840 KB
static constexpr int INV_OFF = 0;                        // 32*9 inverse cells
static constexpr int VOL_OFF = 288;                      // 32 volumes
static constexpr int GS_OFF  = 320;                      // 33 ints (graph starts)
static constexpr int W_OFF   = 384;                      // 32*728 radial weights
static constexpr int SC_OFF  = W_OFF + NGRAPH * NK;          // S_cos [32,728,4]
static constexpr int SS_OFF  = SC_OFF + NGRAPH * NK * NCH;   // S_sin [32,728,4]

__device__ __forceinline__ void klat_of(int kidx, float& ki, float& kj, float& kl) {
    // meshgrid(ij) flat order over [-4,4]^3 with the (0,0,0) entry (flat 364) removed
    int t = kidx + (kidx >= 364 ? 1 : 0);
    ki = (float)(t / 81 - 4);
    kj = (float)((t / 9) % 9 - 4);
    kl = (float)(t % 9 - 4);
}
__device__ __forceinline__ float silu_f(float x) { return x / (1.f + __expf(-x)); }
__device__ __forceinline__ float softplus_f(float x) { return x > 20.f ? x : log1pf(__expf(x)); }

// ---------------------------------------------------------------------------
// Kernel 1: cell inverses, volumes, graph boundary offsets
// ---------------------------------------------------------------------------
__global__ void __launch_bounds__(64) k_setup(const float* __restrict__ cell,
                                              const int* __restrict__ batch,
                                              float* __restrict__ ws) {
    int tid = threadIdx.x;
    float* inv = ws + INV_OFF;
    float* vol = ws + VOL_OFF;
    int*   gs  = (int*)(ws + GS_OFF);
    if (tid < NGRAPH) {
        const float* A = cell + tid * 9;
        float a00=A[0],a01=A[1],a02=A[2],a10=A[3],a11=A[4],a12=A[5],a20=A[6],a21=A[7],a22=A[8];
        float c00 =  (a11*a22 - a12*a21);
        float c01 = -(a10*a22 - a12*a20);
        float c02 =  (a10*a21 - a11*a20);
        float det = a00*c00 + a01*c01 + a02*c02;
        float id  = 1.f / det;
        float* o = inv + tid * 9;  // o[i][j] = (A^-1)[i][j]
        o[0] =  c00*id;                o[1] = -(a01*a22 - a02*a21)*id; o[2] =  (a01*a12 - a02*a11)*id;
        o[3] =  c01*id;                o[4] =  (a00*a22 - a02*a20)*id; o[5] = -(a00*a12 - a02*a10)*id;
        o[6] =  c02*id;                o[7] = -(a00*a21 - a01*a20)*id; o[8] =  (a00*a11 - a01*a10)*id;
        vol[tid] = fmaxf(fabsf(det), KFLOOR);
    } else if (tid < 32 + NGRAPH) {
        int j = tid - 32;                       // lower_bound(batch, j)
        int lo = 0, hi = N_ATOMS;
        while (lo < hi) { int mid = (lo + hi) >> 1; if (batch[mid] < j) lo = mid + 1; else hi = mid; }
        gs[j] = lo;
        if (j == 0) gs[NGRAPH] = N_ATOMS;
    }
}

// ---------------------------------------------------------------------------
// Kernel 2: radial filter MLP; 64x64 layer on WMMA f16 16x16x32.
// 128 threads = 4 wave32s; each wave owns a 16-row tile of the 23296 (b,k) rows.
// ---------------------------------------------------------------------------
__global__ void __launch_bounds__(128) k_filter(const float* __restrict__ W1, const float* __restrict__ b1,
                                                const float* __restrict__ W2, const float* __restrict__ b2,
                                                const float* __restrict__ W3, const float* __restrict__ b3,
                                                float* __restrict__ ws) {
    __shared__ _Float16 w2t[HID * HID];        // W2 transposed: [n][k], 8 KB
    __shared__ _Float16 a_tile[4][16 * HID];   // per-wave layer-1 output (f16), 8 KB
    __shared__ float    h2s[4][16 * HID];      // per-wave layer-2 output, 16 KB
    __shared__ float    skk[4][16];            // per-row |k|

    const int tid = threadIdx.x;
    const int wv  = tid >> 5;
    const int ln  = tid & 31;
    const float* inv = ws + INV_OFF;
    float* wout = ws + W_OFF;

    // stage W2^T into LDS so WMMA B fragments are contiguous per lane
    for (int i = tid; i < HID * HID; i += 128) {
        int n = i >> 6, k = i & 63;
        w2t[i] = (_Float16)W2[k * HID + n];
    }

    const int rowBase = (blockIdx.x * 4 + wv) * 16;   // grid 364 * 4 waves * 16 rows = 23296 exactly

    // layer 1: k-space features -> h1 (f16 A-tile). Each lane fills 2 of 64 cols.
    for (int rr = 0; rr < 16; ++rr) {
        int r = rowBase + rr;
        int b = r / NK, kidx = r % NK;
        float ki, kj, kl; klat_of(kidx, ki, kj, kl);
        const float* ic = inv + b * 9;
        float kx = TWO_PI * (ki*ic[0] + kj*ic[3] + kl*ic[6]);
        float ky = TWO_PI * (ki*ic[1] + kj*ic[4] + kl*ic[7]);
        float kz = TWO_PI * (ki*ic[2] + kj*ic[5] + kl*ic[8]);
        float kn = sqrtf(kx*kx + ky*ky + kz*kz);
        float sk = fmaxf(kn, KFLOOR);
        if (ln == 0) skk[wv][rr] = sk;
        float x0 = log1pf(sk);
        float f0 = x0, f1 = x0 * x0, f2 = 1.f / sk;
        for (int c = ln; c < HID; c += 32) {
            float h = f0 * W1[c] + f1 * W1[HID + c] + f2 * W1[2*HID + c] + b1[c];
            a_tile[wv][rr * HID + c] = (_Float16)silu_f(h);
        }
    }
    __syncthreads();

    // A fragments (16x32 f16 each, K-halves 0..31 and 32..63).
    // ISA layout: lane<16 -> M=lane, K {0..7,16..23}; lane>=16 -> M=lane-16, K {8..15,24..31}.
    v16h afrag[2];
    {
        int rA = ln & 15, koff = (ln >> 4) * 8;
        #pragma unroll
        for (int s = 0; s < 2; ++s) {
            const _Float16* p = &a_tile[wv][rA * HID + s * 32 + koff];
            v8h lo = *(const v8h*)p;
            v8h hi = *(const v8h*)(p + 16);
            v16h a;
            #pragma unroll
            for (int i = 0; i < 8; ++i) { a[i] = lo[i]; a[i + 8] = hi[i]; }
            afrag[s] = a;
        }
    }

    // 4 column tiles x 2 K-steps = 8 v_wmma_f32_16x16x32_f16 per wave.
    // B layout: lane&15 = column N; lane<16 holds K {0..7,16..23}, lane>=16 holds K {8..15,24..31}.
    #pragma unroll
    for (int nt = 0; nt < 4; ++nt) {
        int col = nt * 16 + (ln & 15);
        float bb = b2[col];
        v8f c = {bb, bb, bb, bb, bb, bb, bb, bb};
        #pragma unroll
        for (int s = 0; s < 2; ++s) {
            const _Float16* p = &w2t[col * HID + s * 32 + (ln >> 4) * 8];
            v8h lo = *(const v8h*)p;
            v8h hi = *(const v8h*)(p + 16);
            v16h bfrag;
            #pragma unroll
            for (int i = 0; i < 8; ++i) { bfrag[i] = lo[i]; bfrag[i + 8] = hi[i]; }
            c = __builtin_amdgcn_wmma_f32_16x16x32_f16(false, afrag[s], false, bfrag,
                                                       (short)0, c, false, false);
        }
        // D layout: VGPR g, lane ln -> M = g + 8*(ln>=16), N = ln&15
        int mbase = (ln >> 4) * 8;
        #pragma unroll
        for (int g = 0; g < 8; ++g)
            h2s[wv][(mbase + g) * HID + col] = silu_f(c[g]);
    }
    __syncthreads();

    // layer 3 (64->1) + softplus + 4*pi/k^2 scaling; lanes 0..15 each own a row
    if (ln < 16) {
        int r = rowBase + ln;
        float acc = b3[0];
        #pragma unroll 8
        for (int col = 0; col < HID; ++col)
            acc += h2s[wv][ln * HID + col] * W3[col];
        float scale = softplus_f(acc);
        float sk = skk[wv][ln];
        float wgt = (4.f * PI_F) / (sk * sk) * scale;
        wout[r] = (sk > KFLOOR) ? wgt : 0.f;
    }
}

// ---------------------------------------------------------------------------
// Kernel 3: structure factors. One block per graph; thread t owns k-slots
// {t, t+256, t+512}; accumulate source*cos/sin over the graph's atoms in regs.
// ---------------------------------------------------------------------------
__global__ void __launch_bounds__(256) k_struct(const float* __restrict__ pos,
                                                const float* __restrict__ source,
                                                float* __restrict__ ws) {
    const int b = blockIdx.x;
    const int tid = threadIdx.x;
    const float* inv = ws + INV_OFF + b * 9;
    const int* gs = (const int*)(ws + GS_OFF);
    float* Sc = ws + SC_OFF + b * NK * NCH;
    float* Ss = ws + SS_OFF + b * NK * NCH;

    float i0=inv[0],i1=inv[1],i2=inv[2],i3=inv[3],i4=inv[4],i5=inv[5],i6=inv[6],i7=inv[7],i8=inv[8];

    float ki[3], kj[3], kl[3];
    float sc[3][4] = {{0}}, sn[3][4] = {{0}};
    #pragma unroll
    for (int s = 0; s < 3; ++s) {
        int kidx = tid + s * 256;
        if (kidx < NK) klat_of(kidx, ki[s], kj[s], kl[s]);
        else { ki[s] = kj[s] = kl[s] = 0.f; }
    }

    int n0 = gs[b], n1 = gs[b + 1];
    for (int n = n0; n < n1; ++n) {
        float px = pos[n*3+0], py = pos[n*3+1], pz = pos[n*3+2];
        float f0 = px*i0 + py*i3 + pz*i6;
        float f1 = px*i1 + py*i4 + pz*i7;
        float f2 = px*i2 + py*i5 + pz*i8;
        float q0 = source[n*4+0], q1 = source[n*4+1], q2 = source[n*4+2], q3 = source[n*4+3];
        #pragma unroll
        for (int s = 0; s < 3; ++s) {
            float ph = TWO_PI * (f0*ki[s] + f1*kj[s] + f2*kl[s]);
            float sv, cv; __sincosf(ph, &sv, &cv);
            sc[s][0] += q0*cv; sc[s][1] += q1*cv; sc[s][2] += q2*cv; sc[s][3] += q3*cv;
            sn[s][0] += q0*sv; sn[s][1] += q1*sv; sn[s][2] += q2*sv; sn[s][3] += q3*sv;
        }
    }
    #pragma unroll
    for (int s = 0; s < 3; ++s) {
        int kidx = tid + s * 256;
        if (kidx < NK) {
            #pragma unroll
            for (int c = 0; c < 4; ++c) { Sc[kidx*4 + c] = sc[s][c]; Ss[kidx*4 + c] = sn[s][c]; }
        }
    }
}

// ---------------------------------------------------------------------------
// Kernel 4: per-atom energy. S arrays (745 KB total) stay L2-resident;
// stage is sincos/VALU-bound. Prefetch keeps the float4 S streams ahead.
// ---------------------------------------------------------------------------
__global__ void __launch_bounds__(256) k_energy(const float* __restrict__ pos,
                                                const int* __restrict__ batch,
                                                const float* __restrict__ source,
                                                const float* __restrict__ ws,
                                                float* __restrict__ out) {
    int n = blockIdx.x * 256 + threadIdx.x;     // 32 blocks * 256 = 8192 exactly
    int b = batch[n];
    const float* inv = ws + INV_OFF + b * 9;
    const float4* Sc = (const float4*)(ws + SC_OFF + b * NK * NCH);
    const float4* Ss = (const float4*)(ws + SS_OFF + b * NK * NCH);
    const float* wk  = ws + W_OFF + b * NK;

    float px = pos[n*3], py = pos[n*3+1], pz = pos[n*3+2];
    float f0 = px*inv[0] + py*inv[3] + pz*inv[6];
    float f1 = px*inv[1] + py*inv[4] + pz*inv[7];
    float f2 = px*inv[2] + py*inv[5] + pz*inv[8];
    float q0 = source[n*4], q1 = source[n*4+1], q2 = source[n*4+2], q3 = source[n*4+3];

    float acc = 0.f;
    for (int kidx = 0; kidx < NK; ++kidx) {
        if (kidx + 32 < NK) {                   // global_prefetch_b8 on gfx1250
            __builtin_prefetch(&Sc[kidx + 32], 0, 1);
            __builtin_prefetch(&Ss[kidx + 32], 0, 1);
        }
        float ki, kj, kl; klat_of(kidx, ki, kj, kl);
        float ph = TWO_PI * (f0*ki + f1*kj + f2*kl);
        float sv, cv; __sincosf(ph, &sv, &cv);
        float4 c4 = Sc[kidx];
        float4 s4 = Ss[kidx];
        float dotc = q0*c4.x + q1*c4.y + q2*c4.z + q3*c4.w;
        float dots = q0*s4.x + q1*s4.y + q2*s4.z + q3*s4.w;
        acc += wk[kidx] * (cv*dotc + sv*dots);
    }
    out[n] = 0.5f * acc / ws[VOL_OFF + b];
}

// ---------------------------------------------------------------------------
extern "C" void kernel_launch(void* const* d_in, const int* in_sizes, int n_in,
                              void* d_out, int out_size, void* d_ws, size_t ws_size,
                              hipStream_t stream) {
    const float* pos    = (const float*)d_in[0];
    const int*   batch  = (const int*)  d_in[1];
    const float* cell   = (const float*)d_in[2];
    const float* source = (const float*)d_in[3];
    const float* W1 = (const float*)d_in[4];
    const float* b1 = (const float*)d_in[5];
    const float* W2 = (const float*)d_in[6];
    const float* b2 = (const float*)d_in[7];
    const float* W3 = (const float*)d_in[8];
    const float* b3 = (const float*)d_in[9];
    float* ws  = (float*)d_ws;
    float* out = (float*)d_out;
    (void)in_sizes; (void)n_in; (void)out_size; (void)ws_size;

    hipLaunchKernelGGL(k_setup,  dim3(1),   dim3(64),  0, stream, cell, batch, ws);
    hipLaunchKernelGGL(k_filter, dim3(364), dim3(128), 0, stream, W1, b1, W2, b2, W3, b3, ws);
    hipLaunchKernelGGL(k_struct, dim3(32),  dim3(256), 0, stream, pos, source, ws);
    hipLaunchKernelGGL(k_energy, dim3(32),  dim3(256), 0, stream, pos, batch, source, ws, out);
}
